// GEN_23493471109947
// MI455X (gfx1250) — matrix operand order repaired and checked by reading, not compiled, not run on hardware
//
#include <hip/hip_runtime.h>
#include <hip/hip_bf16.h>

// =====================================================================
// GraphCast-style GNN for MI455X (gfx1250, wave32, WMMA).
//
//  * All 128-wide GEMMs run on v_wmma_f32_16x16x32_bf16 (f32 accum).
//  * Weights converted once to bf16, stored transposed [N=128, K];
//    staged into LDS per block via GLOBAL_LOAD_ASYNC_TO_LDS_B128
//    (ASYNCcnt) — CDNA5 direct global->LDS path, no VGPR round-trip.
//  * First layer of edge/node update fuses the concat-gather
//    (nx[src] | nx[dst] | ef) as per-lane row-pointer selection.
//  * segment_sum fused into edge layer-3 epilogue with non-returning
//    GLOBAL_ATOMIC_ADD_F32 (inline asm, guaranteed native fp atomic).
//  * State carrying residuals (ef, nx, agg) kept fp32; bf16 mirrors
//    feed the next GEMM's A operand.
// =====================================================================

typedef __bf16 bf16;
typedef __attribute__((ext_vector_type(8)))  __bf16 bf16x8;
typedef __attribute__((ext_vector_type(16))) __bf16 bf16x16;
typedef __attribute__((ext_vector_type(8)))  float  f32x8;

#define LATENT     128
#define N_NODES    10242
#define N_EDGES    61440
#define BATCH      2
#define N_TOTAL    (BATCH * N_NODES)   // 20484
#define NUM_ROUNDS 16

__device__ __forceinline__ float silu_f(float x) {
  return x * (1.0f / (1.0f + __expf(-x)));
}

union Frag16 { bf16x16 v; bf16x8 h[2]; };

// Non-returning fp32 global atomic add (GLOBAL_ATOMIC_ADD_F32, STOREcnt).
__device__ __forceinline__ void atomic_add_f32_noret(float* p, float v) {
  asm volatile("global_atomic_add_f32 %0, %1, off"
               : : "v"(p), "v"(v) : "memory");
}

// Async global -> LDS 16B copy (GLOBAL_LOAD_ASYNC_TO_LDS_B128, ASYNCcnt).
// lds_off: byte offset within the workgroup's LDS (addr[31:0] of the
// generic pointer IS the LDS offset per the flat-aperture rules).
__device__ __forceinline__ void async_copy_b128(unsigned lds_off, const void* gaddr) {
  asm volatile("global_load_async_to_lds_b128 %0, %1, off"
               : : "v"(lds_off), "v"(gaddr) : "memory");
}

__device__ __forceinline__ void wait_asynccnt0() {
  asm volatile("s_wait_asynccnt 0x0" ::: "memory");
}

// row-index resolution for concat-gather A operands
// mode: 0 = direct (row=m), 1 = indexed (row=idx[m]),
//       2 = batch broadcast (row = m / N_NODES), 3 = node wrap (row = m % N_NODES)
__device__ __forceinline__ int resolve_row(int mode, int m, const int* __restrict__ idx) {
  switch (mode) {
    case 1:  return idx[m];
    case 2:  return m / N_NODES;
    case 3:  return m % N_NODES;
    default: return m;
  }
}

// ---------------------------------------------------------------------
// Generic fused GEMM:  Y = epilogue( concat(A0,A1,A2)[M x K] @ W[K x 128] + b )
// W passed transposed+bf16: Wt[n*K + k]. K = NSEG*128.
// Each wave: one 16-row tile x all 128 cols (8 wmma accumulators).
// ---------------------------------------------------------------------
template <int NSEG>
__global__ __launch_bounds__(256) void k_gemm(
    const bf16* __restrict__ A0, const bf16* __restrict__ A1, const bf16* __restrict__ A2,
    const int* __restrict__ idx0, const int* __restrict__ idx1,
    int mode0, int mode1, int mode2,
    const bf16* __restrict__ Wt, const float* __restrict__ bias,
    int M, int do_silu,
    bf16* __restrict__ Yb, float* __restrict__ Yf,
    const float* __restrict__ Res, int res_mode,   // 0 none, 1 direct [M,128], 2 latent broadcast
    float* __restrict__ Agg, const int* __restrict__ aggIdx)
{
  constexpr int K = NSEG * 128;
  constexpr int LSTRIDE = K * 2 + 16;   // bytes per N-row in LDS (pad -> bank spread)
  extern __shared__ __align__(16) char smem[];

  // Stage Wt (128 rows x K bf16) into LDS with padded rows using the
  // CDNA5 async global->LDS path. Chunk count is an exact multiple of
  // 256 (K is 128/256/384), so every lane is active on every issue.
  {
    constexpr int CH = K / 8;           // 16B chunks per row
    const unsigned smem_base = (unsigned)(uintptr_t)(&smem[0]);
    for (int v = threadIdx.x; v < 128 * CH; v += 256) {
      int n = v / CH, c = v % CH;
      async_copy_b128(smem_base + n * LSTRIDE + c * 16, (const char*)Wt + v * 16);
    }
    wait_asynccnt0();     // this wave's LDS writes landed
  }
  __syncthreads();        // everyone's writes visible

  const int lane = threadIdx.x & 31;
  const int wave = threadIdx.x >> 5;
  const int half = lane >> 4;
  const int l15  = lane & 15;
  const int tiles = (M + 15) >> 4;

  for (int tile = blockIdx.x * 8 + wave; tile < tiles; tile += gridDim.x * 8) {
    const int mb = tile << 4;
    const int m  = mb + l15;
    const int mc = (m < M) ? m : (M - 1);   // clamp loads on tail tile

    const bf16* __restrict__ rp0 = A0 + (size_t)resolve_row(mode0, mc, idx0) * 128;
    const bf16* rp1 = rp0;
    const bf16* rp2 = rp0;
    if (NSEG >= 2) rp1 = A1 + (size_t)resolve_row(mode1, mc, idx1) * 128;
    if (NSEG >= 3) rp2 = A2 + (size_t)resolve_row(mode2, mc, nullptr) * 128;

    f32x8 acc[8] = {};

#pragma unroll
    for (int s = 0; s < NSEG; ++s) {
      const bf16* __restrict__ ap = (s == 0) ? rp0 : ((s == 1) ? rp1 : rp2);
#pragma unroll
      for (int kk = 0; kk < 128; kk += 32) {
        // A fragment (16x32 bf16): lanes 0-15 hold K kk..kk+7 & kk+16..23,
        // lanes 16-31 hold +8 — two b128 loads per lane.
        Frag16 af;
        af.h[0] = *(const bf16x8*)(ap + kk + half * 8);
        af.h[1] = *(const bf16x8*)(ap + kk + 16 + half * 8);
        const int kc = s * 128 + kk;
#pragma unroll
        for (int nt = 0; nt < 8; ++nt) {
          // B fragment (32x16 bf16): lane (n = l15) holds K kk..kk+7 & kk+16..23,
          // lanes 16-31 hold +8 — matches A layout.
          const char* bp = smem + (nt * 16 + l15) * LSTRIDE + kc * 2 + half * 16;
          Frag16 bfr;
          bfr.h[0] = *(const bf16x8*)(bp);
          bfr.h[1] = *(const bf16x8*)(bp + 32);
          acc[nt] = __builtin_amdgcn_wmma_f32_16x16x32_bf16(
              false, af.v, false, bfr.v, (short)0, acc[nt], false, false);
        }
      }
    }

    // Epilogue. C/D layout: VGPR r, lanes 0-15 -> row r, col lane;
    // lanes 16-31 -> row r+8, col lane-16.
    const int rowbase = mb + half * 8;
#pragma unroll
    for (int nt = 0; nt < 8; ++nt) {
      const int col = nt * 16 + l15;
      const float bv = bias[col];
#pragma unroll
      for (int r = 0; r < 8; ++r) {
        const int row = rowbase + r;
        if (row < M) {
          float v = acc[nt][r] + bv;
          if (do_silu) v = silu_f(v);
          if (res_mode == 1)      v += Res[(size_t)row * 128 + col];
          else if (res_mode == 2) v += Res[(size_t)(row / N_NODES) * 128 + col];
          if (Yf) Yf[(size_t)row * 128 + col] = v;
          if (Yb) Yb[(size_t)row * 128 + col] = (bf16)v;
          if (Agg) atomic_add_f32_noret(&Agg[(size_t)aggIdx[row] * 128 + col], v);
        }
      }
    }
  }
}

// ---------------------------------------------------------------------
// Weight repack: Wt[n*K + k] = bf16(W[k*128 + n])   (W is [K,128] row-major)
// ---------------------------------------------------------------------
__global__ void k_pack_wt(const float* __restrict__ W, bf16* __restrict__ Wt, int K) {
  int t = blockIdx.x * blockDim.x + threadIdx.x;
  if (t >= K * 128) return;
  int n = t / K, k = t % K;
  Wt[t] = (bf16)W[k * 128 + n];
}

// Tiny first layer (K=3 or 4): Y[i,j] = silu(X[i,:]@W[:,j] + b[j]) in bf16
__global__ void k_lin_small(const float* __restrict__ X, int K, int M,
                            const float* __restrict__ W, const float* __restrict__ b,
                            bf16* __restrict__ Y) {
  int t = blockIdx.x * blockDim.x + threadIdx.x;
  if (t >= M * 128) return;
  int i = t >> 7, j = t & 127;
  float s = b[j];
  for (int k = 0; k < K; ++k) s += X[i * K + k] * W[k * 128 + j];
  Y[t] = (bf16)silu_f(s);
}

// Encoder MLP [6->128->128->128] for BATCH=2 rows (one block per row).
__global__ void k_encoder(const float* __restrict__ x,
                          const float* __restrict__ W0, const float* __restrict__ b0,
                          const float* __restrict__ W1, const float* __restrict__ b1,
                          const float* __restrict__ W2, const float* __restrict__ b2,
                          float* __restrict__ latf, bf16* __restrict__ latb) {
  __shared__ float h[128], h2[128];
  int b = blockIdx.x, j = threadIdx.x;
  float s = b0[j];
  for (int k = 0; k < 6; ++k) s += x[b * 6 + k] * W0[k * 128 + j];
  h[j] = silu_f(s);
  __syncthreads();
  s = b1[j];
  for (int k = 0; k < 128; ++k) s += h[k] * W1[k * 128 + j];
  h2[j] = silu_f(s);
  __syncthreads();
  s = b2[j];
  for (int k = 0; k < 128; ++k) s += h2[k] * W2[k * 128 + j];
  latf[b * 128 + j] = s;
  latb[b * 128 + j] = (bf16)s;
}

__global__ void k_zero(float* __restrict__ X, size_t n) {
  size_t t = (size_t)blockIdx.x * blockDim.x + threadIdx.x;
  if (t < n) X[t] = 0.0f;
}

__global__ void k_cvt(const float* __restrict__ X, bf16* __restrict__ Y, size_t n) {
  size_t t = (size_t)blockIdx.x * blockDim.x + threadIdx.x;
  if (t < n) Y[t] = (bf16)X[t];
}

// Mean-pool per graph; partial-sum blocks accumulate atomically into pooled.
__global__ void k_pool(const float* __restrict__ nx, float* __restrict__ pooled) {
  int b = blockIdx.x >> 3, slice = blockIdx.x & 7, j = threadIdx.x;
  int i0 = slice * (N_NODES / 8);
  int i1 = (slice == 7) ? N_NODES : i0 + (N_NODES / 8);
  float s = 0.0f;
  for (int i = i0; i < i1; ++i)
    s += nx[((size_t)(b * N_NODES + i)) * 128 + j];
  atomic_add_f32_noret(&pooled[b * 128 + j], s * (1.0f / (float)N_NODES));
}

// Decoder MLP [128 -> 128 (silu) -> 1]
__global__ void k_dec(const float* __restrict__ pooled,
                      const float* __restrict__ W0, const float* __restrict__ b0,
                      const float* __restrict__ W1, const float* __restrict__ b1,
                      float* __restrict__ out) {
  __shared__ float h[128], red[128];
  int b = blockIdx.x, j = threadIdx.x;
  float s = b0[j];
  for (int k = 0; k < 128; ++k) s += pooled[b * 128 + k] * W0[k * 128 + j];
  h[j] = silu_f(s);
  __syncthreads();
  red[j] = h[j] * W1[j];
  __syncthreads();
  for (int off = 64; off > 0; off >>= 1) {
    if (j < off) red[j] += red[j + off];
    __syncthreads();
  }
  if (j == 0) out[b] = red[0] + b1[0];
}

// ---------------------------------------------------------------------
// Host orchestration
// ---------------------------------------------------------------------
extern "C" void kernel_launch(void* const* d_in, const int* in_sizes, int n_in,
                              void* d_out, int out_size, void* d_ws, size_t ws_size,
                              hipStream_t stream) {
  (void)in_sizes; (void)n_in; (void)out_size; (void)ws_size;

  const float* x         = (const float*)d_in[0];
  const float* node_pos  = (const float*)d_in[1];
  const int*   srcIdx    = (const int*)d_in[2];
  const int*   dstIdx    = ((const int*)d_in[2]) + N_EDGES;
  const float* edge_feat = (const float*)d_in[3];
  auto P = [&](int i) { return (const float*)d_in[i]; };
  // param bases: enc=4, pos=10, node=16, emap=22, eupd=28, nupd=34, dec=40

  // ---- workspace carve (all 256B aligned) ----
  char* wp = (char*)d_ws;
  auto carve = [&](size_t bytes) -> char* {
    char* r = wp; wp += (bytes + 255) & ~(size_t)255; return r;
  };
  bf16*  wtPos1 = (bf16*)carve(128 * 128 * sizeof(bf16));
  bf16*  wtPos2 = (bf16*)carve(128 * 128 * sizeof(bf16));
  bf16*  wtNod0 = (bf16*)carve(128 * 256 * sizeof(bf16));
  bf16*  wtNod1 = (bf16*)carve(128 * 128 * sizeof(bf16));
  bf16*  wtNod2 = (bf16*)carve(128 * 128 * sizeof(bf16));
  bf16*  wtEmp1 = (bf16*)carve(128 * 128 * sizeof(bf16));
  bf16*  wtEmp2 = (bf16*)carve(128 * 128 * sizeof(bf16));
  bf16*  wtEup0 = (bf16*)carve(128 * 384 * sizeof(bf16));
  bf16*  wtEup1 = (bf16*)carve(128 * 128 * sizeof(bf16));
  bf16*  wtEup2 = (bf16*)carve(128 * 128 * sizeof(bf16));
  bf16*  wtNup0 = (bf16*)carve(128 * 256 * sizeof(bf16));
  bf16*  wtNup1 = (bf16*)carve(128 * 128 * sizeof(bf16));
  bf16*  wtNup2 = (bf16*)carve(128 * 128 * sizeof(bf16));
  float* latf   = (float*)carve(BATCH * 128 * sizeof(float));
  bf16*  latb   = (bf16*) carve(BATCH * 128 * sizeof(bf16));
  bf16*  pemb   = (bf16*) carve((size_t)N_NODES * 128 * sizeof(bf16));
  bf16*  hA     = (bf16*) carve((size_t)N_EDGES * 128 * sizeof(bf16));
  bf16*  hB     = (bf16*) carve((size_t)N_EDGES * 128 * sizeof(bf16));
  float* ef     = (float*)carve((size_t)N_EDGES * 128 * sizeof(float));
  bf16*  efb    = (bf16*) carve((size_t)N_EDGES * 128 * sizeof(bf16));
  float* nx     = (float*)carve((size_t)N_TOTAL * 128 * sizeof(float));
  bf16*  nxb    = (bf16*) carve((size_t)N_TOTAL * 128 * sizeof(bf16));
  float* agg    = (float*)carve((size_t)N_TOTAL * 128 * sizeof(float));
  bf16*  aggb   = (bf16*) carve((size_t)N_TOTAL * 128 * sizeof(bf16));
  float* pooled = (float*)carve(BATCH * 128 * sizeof(float));

  // Allow big dynamic LDS for the GEMM kernels (ignore errors if no-op on AMD)
  (void)hipFuncSetAttribute((const void*)k_gemm<1>, hipFuncAttributeMaxDynamicSharedMemorySize, 128 * (1 * 256 + 16));
  (void)hipFuncSetAttribute((const void*)k_gemm<2>, hipFuncAttributeMaxDynamicSharedMemorySize, 128 * (2 * 256 + 16));
  (void)hipFuncSetAttribute((const void*)k_gemm<3>, hipFuncAttributeMaxDynamicSharedMemorySize, 128 * (3 * 256 + 16));

  auto pack = [&](const float* W, bf16* Wt, int K) {
    k_pack_wt<<<(K * 128 + 255) / 256, 256, 0, stream>>>(W, Wt, K);
  };
  pack(P(12), wtPos1, 128); pack(P(14), wtPos2, 128);
  pack(P(16), wtNod0, 256); pack(P(18), wtNod1, 128); pack(P(20), wtNod2, 128);
  pack(P(24), wtEmp1, 128); pack(P(26), wtEmp2, 128);
  pack(P(28), wtEup0, 384); pack(P(30), wtEup1, 128); pack(P(32), wtEup2, 128);
  pack(P(34), wtNup0, 256); pack(P(36), wtNup1, 128); pack(P(38), wtNup2, 128);

  auto gemm = [&](int nseg, int M,
                  const bf16* A0, const bf16* A1, const bf16* A2,
                  const int* i0, const int* i1, int m0, int m1, int m2,
                  const bf16* Wt, const float* bias, int do_silu,
                  bf16* Yb, float* Yf, const float* Res, int res_mode,
                  float* Agg, const int* aggI) {
    int tiles  = (M + 15) / 16;
    int blocks = (tiles + 7) / 8;
    size_t sh  = (size_t)128 * (nseg * 256 + 16);
    switch (nseg) {
      case 1: k_gemm<1><<<blocks, 256, sh, stream>>>(A0, A1, A2, i0, i1, m0, m1, m2, Wt, bias, M, do_silu, Yb, Yf, Res, res_mode, Agg, aggI); break;
      case 2: k_gemm<2><<<blocks, 256, sh, stream>>>(A0, A1, A2, i0, i1, m0, m1, m2, Wt, bias, M, do_silu, Yb, Yf, Res, res_mode, Agg, aggI); break;
      case 3: k_gemm<3><<<blocks, 256, sh, stream>>>(A0, A1, A2, i0, i1, m0, m1, m2, Wt, bias, M, do_silu, Yb, Yf, Res, res_mode, Agg, aggI); break;
    }
  };

  // ---- encoder: latent [2,128] ----
  k_encoder<<<BATCH, 128, 0, stream>>>(x, P(4), P(5), P(6), P(7), P(8), P(9), latf, latb);

  // ---- pos_mlp: node_pos [N,3] -> pos_emb bf16 [N,128] ----
  k_lin_small<<<(N_NODES * 128 + 255) / 256, 256, 0, stream>>>(node_pos, 3, N_NODES, P(10), P(11), hA);
  gemm(1, N_NODES, hA, nullptr, nullptr, nullptr, nullptr, 0, 0, 0, wtPos1, P(13), 1, hB, nullptr, nullptr, 0, nullptr, nullptr);
  gemm(1, N_NODES, hB, nullptr, nullptr, nullptr, nullptr, 0, 0, 0, wtPos2, P(15), 0, pemb, nullptr, nullptr, 0, nullptr, nullptr);

  // ---- node init: node_mlp(concat(latent bcast, pos_emb)) + latent ----
  gemm(2, N_TOTAL, latb, pemb, nullptr, nullptr, nullptr, 2, 3, 0, wtNod0, P(17), 1, hA, nullptr, nullptr, 0, nullptr, nullptr);
  gemm(1, N_TOTAL, hA, nullptr, nullptr, nullptr, nullptr, 0, 0, 0, wtNod1, P(19), 1, hB, nullptr, nullptr, 0, nullptr, nullptr);
  gemm(1, N_TOTAL, hB, nullptr, nullptr, nullptr, nullptr, 0, 0, 0, wtNod2, P(21), 0, nxb, nx, latf, 2, nullptr, nullptr);

  // ---- edge feature map: ef [E,128] ----
  k_lin_small<<<(N_EDGES * 128 + 255) / 256, 256, 0, stream>>>(edge_feat, 4, N_EDGES, P(22), P(23), hA);
  gemm(1, N_EDGES, hA, nullptr, nullptr, nullptr, nullptr, 0, 0, 0, wtEmp1, P(25), 1, hB, nullptr, nullptr, 0, nullptr, nullptr);
  gemm(1, N_EDGES, hB, nullptr, nullptr, nullptr, nullptr, 0, 0, 0, wtEmp2, P(27), 0, efb, ef, nullptr, 0, nullptr, nullptr);

  // ---- 16 message-passing rounds (shared weights) ----
  const size_t nAgg = (size_t)N_TOTAL * 128;
  for (int r = 0; r < NUM_ROUNDS; ++r) {
    k_zero<<<(int)((nAgg + 255) / 256), 256, 0, stream>>>(agg, nAgg);
    // edge update: MLP(concat(nx[src], nx[dst], ef)) + ef; scatter-add into agg[dst]
    gemm(3, N_EDGES, nxb, nxb, efb, srcIdx, dstIdx, 1, 1, 0, wtEup0, P(29), 1, hA, nullptr, nullptr, 0, nullptr, nullptr);
    gemm(1, N_EDGES, hA, nullptr, nullptr, nullptr, nullptr, 0, 0, 0, wtEup1, P(31), 1, hB, nullptr, nullptr, 0, nullptr, nullptr);
    gemm(1, N_EDGES, hB, nullptr, nullptr, nullptr, nullptr, 0, 0, 0, wtEup2, P(33), 0, efb, ef, ef, 1, agg, dstIdx);
    k_cvt<<<(int)((nAgg + 255) / 256), 256, 0, stream>>>(agg, aggb, nAgg);
    // node update: MLP(concat(nx, agg)) + nx
    gemm(2, N_TOTAL, nxb, aggb, nullptr, nullptr, nullptr, 0, 0, 0, wtNup0, P(35), 1, hA, nullptr, nullptr, 0, nullptr, nullptr);
    gemm(1, N_TOTAL, hA, nullptr, nullptr, nullptr, nullptr, 0, 0, 0, wtNup1, P(37), 1, hB, nullptr, nullptr, 0, nullptr, nullptr);
    gemm(1, N_TOTAL, hB, nullptr, nullptr, nullptr, nullptr, 0, 0, 0, wtNup2, P(39), 0, nxb, nx, nx, 1, nullptr, nullptr);
  }

  // ---- pool + decode ----
  k_zero<<<1, 256, 0, stream>>>(pooled, BATCH * 128);
  k_pool<<<BATCH * 8, 128, 0, stream>>>(nx, pooled);
  k_dec<<<BATCH, 128, 0, stream>>>(pooled, P(40), P(41), P(42), P(43), (float*)d_out);
}
